// GraphNN_KNN_v1_75368086110793
// MI455X (gfx1250) — matrix operand order repaired and checked by reading, not compiled, not run on hardware
//
#include <hip/hip_runtime.h>

typedef __attribute__((ext_vector_type(16))) _Float16 v16h;
typedef __attribute__((ext_vector_type(8)))  _Float16 h8;
typedef __attribute__((ext_vector_type(8)))  float    v8f;

#define H 64
#define WAVES_PER_BLOCK 8

// ---------------------------------------------------------------------------
// Small helper kernels
// ---------------------------------------------------------------------------

// Pack fp32 weights [rows][64] into f16 in WMMA-B fragment order:
//   out[((ks*4 + nt)*32 + lane)*16 + t] = W[ks*32 + (lane>>4)*16 + t][nt*16 + (lane&15)]
// (zero-padded to kpad rows). Each lane's 16 halves for one 16x16x32 WMMA B
// fragment are then 32 contiguous bytes in LDS -> two ds_load_b128.
__global__ void pack_w_f16_swz_kernel(const float* __restrict__ w, _Float16* __restrict__ out,
                                      int rows, int kpad, int cols) {
  int i = blockIdx.x * blockDim.x + threadIdx.x;
  int total = kpad * cols;
  if (i >= total) return;
  int t    = i & 15;
  int lane = (i >> 4) & 31;
  int nt   = (i >> 9) & 3;
  int ks   = i >> 11;
  int k = ks * 32 + (lane >> 4) * 16 + t;
  int c = nt * 16 + (lane & 15);
  out[i] = (k < rows) ? (_Float16)w[k * cols + c] : (_Float16)0.0f;
}

__global__ void lin_relu_kernel(const float* __restrict__ x, const float* __restrict__ w,
                                const float* __restrict__ b, float* __restrict__ y,
                                int n, int din, int dout) {
  int i = blockIdx.x * blockDim.x + threadIdx.x;
  if (i >= n * dout) return;
  int row = i / dout, o = i % dout;
  float acc = b[o];
  for (int k = 0; k < din; ++k) acc += x[(size_t)row * din + k] * w[k * dout + o];
  y[i] = acc > 0.f ? acc : 0.f;
}

__global__ void f32_to_f16_kernel(const float* __restrict__ x, _Float16* __restrict__ y, int n) {
  int i = blockIdx.x * blockDim.x + threadIdx.x;
  if (i < n) y[i] = (_Float16)x[i];
}

__global__ void zero_kernel(float* __restrict__ p, int n) {
  int i = blockIdx.x * blockDim.x + threadIdx.x;
  if (i < n) p[i] = 0.f;
}

__global__ void update_avg_kernel(float* __restrict__ x, const float* __restrict__ agg, int n) {
  int i = blockIdx.x * blockDim.x + threadIdx.x;
  if (i < n) x[i] = (x[i] + agg[i]) * 0.5f;
}

__global__ void out_kernel(const float* __restrict__ x, const float* __restrict__ w,
                           const float* __restrict__ b, float* __restrict__ y,
                           int n, int dout) {
  int i = blockIdx.x * blockDim.x + threadIdx.x;
  if (i >= n * dout) return;
  int row = i / dout, o = i % dout;
  float acc = b[o];
  for (int k = 0; k < H; ++k) acc += x[(size_t)row * H + k] * w[k * dout + o];
  y[i] = acc;
}

// ---------------------------------------------------------------------------
// Edge-message WMMA kernel.
//   msg = relu([ x[a] | x[b]-x[a] | ef? ] @ W + bias)   -> scatter into agg[g]
//   One wave per 16-edge tile. A (16 x KPAD f16) staged in LDS from gathered
//   node rows; B (KPAD x 64 f16) pre-swizzled to fragment order in LDS.
//   C = 16x64 f32 accumulated with v_wmma_f32_16x16x32_f16.
//   MAXAGG=false -> atomicAdd (EmulsionConv, aggregate at src)
//   MAXAGG=true  -> int-punned atomicMax (EdgeConv; relu >= 0 so integer
//                   ordering == float ordering; 0-init == empty-segment 0)
// ---------------------------------------------------------------------------
template <int KPAD, bool MAXAGG>
__global__ __launch_bounds__(WAVES_PER_BLOCK * 32)
void edge_msg_kernel(const _Float16* __restrict__ xh,
                     const int* __restrict__ a_idx,     // rows for x_i
                     const int* __restrict__ b_idx,     // rows for x_j
                     const int* __restrict__ g_idx,     // aggregation rows
                     const float* __restrict__ ef,      // edge features (or null)
                     const _Float16* __restrict__ w,    // swizzled [KPAD/32][4][32][16] f16
                     const float* __restrict__ bias,    // [64]
                     float* __restrict__ agg,           // [N][64]
                     int e_begin, int n_tiles) {
  __shared__ _Float16 s_w[KPAD * H];                    // fragment-ordered weights
  __shared__ _Float16 s_st[WAVES_PER_BLOCK * 16 * KPAD];

  for (int i = threadIdx.x; i < KPAD * H; i += blockDim.x) s_w[i] = w[i];
  __syncthreads();

  const int lane = threadIdx.x & 31;
  const int wave = threadIdx.x >> 5;
  _Float16* st = s_st + wave * 16 * KPAD;
  const int m16 = lane & 15;   // A row / C-N column within half-wave
  const int hh  = lane >> 4;   // half-wave select

  int tile = blockIdx.x * WAVES_PER_BLOCK + wave;
  const int stride = gridDim.x * WAVES_PER_BLOCK;

  for (; tile < n_tiles; tile += stride) {
    const int eb = e_begin + tile * 16;

    // ---- stage A tile: 2 lanes per edge, each covers 32 feature columns ----
    {
      int e = lane >> 1;
      int c0 = (lane & 1) * 32;
      int ai = a_idx[eb + e];
      int bi = b_idx[eb + e];
      const h8* xa = (const h8*)(xh + (size_t)ai * H + c0);
      const h8* xb = (const h8*)(xh + (size_t)bi * H + c0);
      _Float16* row = st + e * KPAD;
      h8* rw0 = (h8*)(row + c0);       // x_i block
      h8* rw1 = (h8*)(row + H + c0);   // (x_j - x_i) block
#pragma unroll
      for (int q = 0; q < 4; ++q) {
        h8 xi = xa[q];
        h8 xj = xb[q];
        rw0[q] = xi;
        rw1[q] = xj - xi;
      }
      if constexpr (KPAD > 2 * H) {    // edge feature + zero pad to KPAD
        h8 z = {};
        if (c0 == 0) {
          h8 efv = {};
          efv[0] = (_Float16)ef[eb + e];
          *(h8*)(row + 2 * H)     = efv;   // cols 128..135 (ef, zeros)
          *(h8*)(row + 2 * H + 8) = z;     // cols 136..143
        } else {
          *(h8*)(row + 2 * H + 16) = z;    // cols 144..151
          *(h8*)(row + 2 * H + 24) = z;    // cols 152..159
        }
      }
    }
    asm volatile("s_wait_dscnt 0x0" ::: "memory");  // wave-local LDS visibility

    // scatter element offsets (32-bit, non-negative -> saddr+voffset atomics)
    // for the 8 C-rows this lane owns (rows hh*8 .. hh*8+7)
    int goff[8];
#pragma unroll
    for (int r = 0; r < 8; ++r) goff[r] = g_idx[eb + r + 8 * hh] * H;

#pragma unroll
    for (int nt = 0; nt < 4; ++nt) {
      v8f acc = {};
#pragma unroll
      for (int ks = 0; ks < KPAD / 32; ++ks) {
        // A fragment: lane m16 = row M, halves K = ks*32 + 16*hh + {0..15}
        const h8* ap = (const h8*)(st + m16 * KPAD + ks * 32 + 16 * hh);
        h8 alo = ap[0];
        h8 ahi = ap[1];
        v16h a = __builtin_shufflevector(alo, ahi, 0, 1, 2, 3, 4, 5, 6, 7,
                                         8, 9, 10, 11, 12, 13, 14, 15);
        // B fragment: pre-swizzled, 32 contiguous bytes per lane
        const h8* bp = (const h8*)(s_w + ((((ks * 4 + nt) * 32) + lane) << 4));
        h8 blo = bp[0];
        h8 bhi = bp[1];
        v16h b = __builtin_shufflevector(blo, bhi, 0, 1, 2, 3, 4, 5, 6, 7,
                                         8, 9, 10, 11, 12, 13, 14, 15);
        acc = __builtin_amdgcn_wmma_f32_16x16x32_f16(
            false, a, false, b, (short)0, acc, false, false);
      }
      const int n = nt * 16 + m16;
      const float bv = bias[n];
#pragma unroll
      for (int r = 0; r < 8; ++r) {
        float val = acc[r] + bv;
        val = val > 0.f ? val : 0.f;           // relu
        int off = goff[r] + n;                 // 32-bit element offset
        __builtin_assume(off >= 0);
        if constexpr (MAXAGG) {
          atomicMax((int*)agg + off, __float_as_int(val));
        } else {
          atomicAdd(agg + off, val);
        }
      }
    }
  }
}

// ---------------------------------------------------------------------------
// Host orchestration
// ---------------------------------------------------------------------------
extern "C" void kernel_launch(void* const* d_in, const int* in_sizes, int n_in,
                              void* d_out, int out_size, void* d_ws, size_t ws_size,
                              hipStream_t stream) {
  (void)n_in; (void)out_size; (void)ws_size;

  const float* x0    = (const float*)d_in[0];
  const int*   eidx  = (const int*)d_in[1];
  const float* ef    = (const float*)d_in[2];
  const float* lin_w[3] = {(const float*)d_in[3], (const float*)d_in[5], (const float*)d_in[7]};
  const float* lin_b[3] = {(const float*)d_in[4], (const float*)d_in[6], (const float*)d_in[8]};
  const float* em_w  = (const float*)d_in[9];    // [3][129][64]
  const float* em_b  = (const float*)d_in[10];   // [3][64]
  const float* ec_w  = (const float*)d_in[11];   // [3][128][64]
  const float* ec_b  = (const float*)d_in[12];   // [3][64]
  const float* ow    = (const float*)d_in[13];   // [64][10]
  const float* ob    = (const float*)d_in[14];   // [10]

  const int D_IN = 10, D_OUT = 10;
  const int N = in_sizes[0] / D_IN;
  const int E = in_sizes[1] / 2;
  const int per = E / 4;                         // N_ORDERS = 4
  const int EMK = 2 * H + 1;                     // 129
  const int EMKP = 160;                          // padded to 5 x 32
  const int ECK = 2 * H;                         // 128

  const int* src = eidx;          // edge_index[0]
  const int* dst = eidx + E;      // edge_index[1]

  // workspace carve-out
  char* p = (char*)d_ws;
  float* xA = (float*)p;      p += (size_t)N * H * sizeof(float);
  float* xB = (float*)p;      p += (size_t)N * H * sizeof(float);
  float* xC = (float*)p;      p += (size_t)N * H * sizeof(float);
  _Float16* xh   = (_Float16*)p; p += (size_t)N * H * sizeof(_Float16);
  _Float16* emwh = (_Float16*)p; p += (size_t)3 * EMKP * H * sizeof(_Float16);
  _Float16* ecwh = (_Float16*)p; p += (size_t)3 * ECK * H * sizeof(_Float16);

  float* cur  = xA;
  float* tmp  = xB;
  float* aggp = xC;

  // pack + swizzle weights fp32 -> f16 (em padded 129->160 rows)
  {
    dim3 g1((EMKP * H + 255) / 256), g2((ECK * H + 255) / 256);
    for (int i = 0; i < 3; ++i)
      pack_w_f16_swz_kernel<<<g1, 256, 0, stream>>>(em_w + (size_t)i * EMK * H,
                                                    emwh + (size_t)i * EMKP * H, EMK, EMKP, H);
    for (int i = 0; i < 3; ++i)
      pack_w_f16_swz_kernel<<<g2, 256, 0, stream>>>(ec_w + (size_t)i * ECK * H,
                                                    ecwh + (size_t)i * ECK * H, ECK, ECK, H);
  }

  const int elems = N * H;
  dim3 gE((elems + 255) / 256);

  // --- 3 x (linear + EmulsionConv) ---
  for (int L = 0; L < 3; ++L) {
    const float* lin_in = (L == 0) ? x0 : cur;
    const int din = (L == 0) ? D_IN : H;
    lin_relu_kernel<<<gE, 256, 0, stream>>>(lin_in, lin_w[L], lin_b[L], tmp, N, din, H);
    { float* t = cur; cur = tmp; tmp = t; }

    const int tiles = per / 16;
    const dim3 gM((tiles + WAVES_PER_BLOCK - 1) / WAVES_PER_BLOCK);
    for (int o = 0; o < 4; ++o) {
      f32_to_f16_kernel<<<gE, 256, 0, stream>>>(cur, xh, elems);
      zero_kernel<<<gE, 256, 0, stream>>>(aggp, elems);
      // x_i = x[dst], x_j = x[src], aggregate sum at src
      edge_msg_kernel<160, false><<<gM, WAVES_PER_BLOCK * 32, 0, stream>>>(
          xh, dst, src, src, ef,
          emwh + (size_t)L * EMKP * H, em_b + L * H, aggp, o * per, tiles);
      update_avg_kernel<<<gE, 256, 0, stream>>>(cur, aggp, elems);
    }
  }

  // --- 3 x EdgeConv (max aggregation at dst) ---
  {
    const int tiles = E / 16;
    const dim3 gM((tiles + WAVES_PER_BLOCK - 1) / WAVES_PER_BLOCK);
    for (int L = 0; L < 3; ++L) {
      f32_to_f16_kernel<<<gE, 256, 0, stream>>>(cur, xh, elems);
      zero_kernel<<<gE, 256, 0, stream>>>(aggp, elems);
      edge_msg_kernel<128, true><<<gM, WAVES_PER_BLOCK * 32, 0, stream>>>(
          xh, dst, src, dst, nullptr,
          ecwh + (size_t)L * ECK * H, ec_b + L * H, aggp, 0, tiles);
      { float* t = cur; cur = aggp; aggp = t; }  // x <- agg
    }
  }

  // --- output projection 64 -> 10 ---
  out_kernel<<<((N * D_OUT) + 255) / 256, 256, 0, stream>>>(cur, ow, ob, (float*)d_out, N, D_OUT);
}
